// VOXCPM_FEAT_ENCODER_403726926218
// MI455X (gfx1250) — matrix operand order repaired and checked by reading, not compiled, not run on hardware
//
#include <hip/hip_runtime.h>
#include <hip/hip_bf16.h>
#include <stdint.h>

typedef __bf16 bf16;
typedef __attribute__((ext_vector_type(16))) __bf16 bf16x16;
typedef __attribute__((ext_vector_type(8)))  float  f32x8;
typedef __attribute__((ext_vector_type(4)))  unsigned int u32x4;
typedef __attribute__((ext_vector_type(8)))  int i32x8;
typedef __attribute__((ext_vector_type(4)))  int i32x4;

#define N_SEQ   8192
#define P_AUD   8
#define S_LEN   9
#define FEATD   64
#define DMODEL  1024
#define KVH     2
#define G_HEADS 8
#define HD      64
#define HHALF   32
#define IFF     4096
#define LAYERS  8
#define LMV     2048
#define DQKV    1280                 // D + 2*KV*HD
#define MTOK    (N_SEQ * S_LEN)     // 73728

#if defined(__has_builtin)
#if __has_builtin(__builtin_amdgcn_tensor_load_to_lds)
#define HAVE_TDM 1
#endif
#endif

// ---------------------------------------------------------------------------
// fp32 -> bf16 cast (weights, done once per launch)
// ---------------------------------------------------------------------------
__global__ void f2bf_kernel(const float* __restrict__ src, bf16* __restrict__ dst, size_t n) {
    size_t i = (size_t)blockIdx.x * blockDim.x + threadIdx.x;
    size_t stride = (size_t)gridDim.x * blockDim.x;
    for (; i < n; i += stride) dst[i] = (bf16)src[i];
}

// ---------------------------------------------------------------------------
// in_proj: h[n,p,:] = audio[n,p,:] @ W^T + b   (K = 64, negligible FLOPs)
// ---------------------------------------------------------------------------
__global__ void in_proj_kernel(const float* __restrict__ audio, const float* __restrict__ w,
                               const float* __restrict__ b, float* __restrict__ H) {
    __shared__ float sa[FEATD];
    int r = blockIdx.x;                    // 0 .. N*P-1
    int n = r / P_AUD, p = r % P_AUD;
    int tid = threadIdx.x;
    if (tid < FEATD) sa[tid] = audio[(size_t)r * FEATD + tid];
    __syncthreads();
    for (int d = tid; d < DMODEL; d += blockDim.x) {
        float acc = b[d];
        const float* wr = w + (size_t)d * FEATD;
#pragma unroll 8
        for (int k = 0; k < FEATD; ++k) acc += sa[k] * wr[k];
        H[((size_t)n * S_LEN + 1 + p) * DMODEL + d] = acc;
    }
}

__global__ void copy_special_kernel(const float* __restrict__ sp, float* __restrict__ H) {
    size_t i = (size_t)blockIdx.x * blockDim.x + threadIdx.x;
    size_t stride = (size_t)gridDim.x * blockDim.x;
    size_t total = (size_t)N_SEQ * DMODEL;
    for (; i < total; i += stride) {
        size_t n = i >> 10, d = i & 1023;
        H[n * S_LEN * DMODEL + d] = sp[i];
    }
}

// ---------------------------------------------------------------------------
// rmsnorm (reference: x * rsqrt(sum(x*x)), no eps) -> bf16 rows
// ---------------------------------------------------------------------------
__global__ void rmsnorm_kernel(const float* __restrict__ in, size_t inRowStride,
                               bf16* __restrict__ out) {
    int r = blockIdx.x;
    int tid = threadIdx.x;
    const float* x = in + (size_t)r * inRowStride;
    float vals[4];
    float s = 0.f;
#pragma unroll
    for (int i = 0; i < 4; ++i) { vals[i] = x[tid + i * 256]; s += vals[i] * vals[i]; }
#pragma unroll
    for (int off = 16; off > 0; off >>= 1) s += __shfl_xor(s, off, 32);
    __shared__ float wsum[8];
    if ((tid & 31) == 0) wsum[tid >> 5] = s;
    __syncthreads();
    float tot = 0.f;
#pragma unroll
    for (int i = 0; i < 8; ++i) tot += wsum[i];
    float scale = rsqrtf(tot);
    bf16* o = out + (size_t)r * DMODEL;
#pragma unroll
    for (int i = 0; i < 4; ++i) o[tid + i * 256] = (bf16)(vals[i] * scale);
}

// ---------------------------------------------------------------------------
// TDM: one 2D-tile DMA (rows x 32 bf16) global -> LDS with descriptor-driven
// padding (8 DWORDs after every 16 DWORDs) => LDS row stride = 48 halves.
// ---------------------------------------------------------------------------
#define BM 128
#define BN 64
#define BK 32
#define LDT 48

#ifdef HAVE_TDM
__device__ __forceinline__ void tdm_load_tile(const bf16* gsrc, unsigned lds_off,
                                              int rows, int srcStride) {
    unsigned long long ga = (unsigned long long)(uintptr_t)gsrc;
    u32x4 g0;
    g0[0] = 1u;                                                 // count=1 valid, user mode
    g0[1] = lds_off;                                            // lds_addr
    g0[2] = (unsigned)(ga & 0xFFFFFFFFu);                       // global_addr[31:0]
    g0[3] = (unsigned)((ga >> 32) & 0x1FFFFFFu) | (2u << 30);   // global_addr[56:32], type=2
    unsigned dim0 = (unsigned)srcStride;                        // tensor width (elements)
    unsigned dim1 = (unsigned)rows;
    i32x8 g1;
    g1[0] = (int)((1u << 16) |            // data_size = 2 bytes
                  (1u << 20) |            // pad_enable
                  (3u << 22) |            // pad_interval: 16 DWORDs
                  (7u << 25));            // pad_amount: 8 DWORDs
    g1[1] = (int)((dim0 & 0xFFFFu) << 16);                              // tensor_dim0 lo
    g1[2] = (int)(((dim0 >> 16) & 0xFFFFu) | ((dim1 & 0xFFFFu) << 16)); // dim0 hi | dim1 lo
    g1[3] = (int)(((dim1 >> 16) & 0xFFFFu) | ((unsigned)BK << 16));     // dim1 hi | tile_dim0
    g1[4] = rows & 0xFFFF;                                              // tile_dim1
    g1[5] = srcStride;                                                  // dim0_stride lo32
    g1[6] = 0;
    g1[7] = 0;
    i32x4 g2 = {0, 0, 0, 0};
    i32x4 g3 = {0, 0, 0, 0};
#if __clang_major__ >= 23
    i32x8 g4 = {0, 0, 0, 0, 0, 0, 0, 0};
    __builtin_amdgcn_tensor_load_to_lds(g0, g1, g2, g3, g4, 0);
#else
    __builtin_amdgcn_tensor_load_to_lds(g0, g1, g2, g3, 0);
#endif
}
#endif

// ---------------------------------------------------------------------------
// Templated bf16 WMMA GEMM:  C[M,N] = A[M,K] * B[N,K]^T  (fp32 accumulate)
// block tile 128x64, BK=32, 8 waves -> 32x32 per wave (2x2 WMMA frags).
// TDM path: double-buffered LDS, wave 0 DMAs tile k+1 while all compute k.
// MODE 0: store fp32        MODE 1: store bf16
// MODE 2: fp32 += residual  MODE 3: fused gate/up: out = silu(C)*C2 (bf16)
// ---------------------------------------------------------------------------
template <int MODE>
__global__ __launch_bounds__(256) void gemm_bf16_wmma(
    const bf16* __restrict__ A, const bf16* __restrict__ B,
    float* __restrict__ Cf, bf16* __restrict__ Cb,
    int M, int N, int K) {
    constexpr bool FUSED = (MODE == 3);
    __shared__ bf16 lA[2 * BM * LDT];
    __shared__ bf16 lB[2 * BN * LDT];
    __shared__ bf16 lB2[FUSED ? 2 * BN * LDT : 16];
    int tid = threadIdx.x;
    int lane = tid & 31;
    int wid = tid >> 5;
    int waveM = wid & 3;        // 4 waves along M
    int waveN = wid >> 2;       // 2 waves along N
    int rowBase = blockIdx.y * BM;
    int colBase = blockIdx.x * BN;
    int lhalf = lane & 15;
    int hi = lane >> 4;

    f32x8 acc[2][2] = {};
    f32x8 acc2[2][2] = {};

    union Frag { bf16x16 v; u32x4 q[2]; };

#ifdef HAVE_TDM
    unsigned ldsA_off  = (unsigned)(uintptr_t)(void*)lA;
    unsigned ldsB_off  = (unsigned)(uintptr_t)(void*)lB;
    unsigned ldsB2_off = (unsigned)(uintptr_t)(void*)lB2;
    const unsigned abuf = BM * LDT * 2;   // bytes per A buffer
    const unsigned bbuf = BN * LDT * 2;   // bytes per B buffer

    // prologue: DMA tile 0 into buffer 0
    if (wid == 0) {
        tdm_load_tile(A + (size_t)rowBase * K, ldsA_off, BM, K);
        tdm_load_tile(B + (size_t)colBase * K, ldsB_off, BN, K);
        if (FUSED)
            tdm_load_tile(B + (size_t)(IFF + colBase) * K, ldsB2_off, BN, K);
        __builtin_amdgcn_s_wait_tensorcnt(0);
    }
    __syncthreads();

    int cur = 0;
    for (int k0 = 0; k0 < K; k0 += BK) {
        // issue DMA for next tile into the alternate buffer (overlap with WMMA)
        if (wid == 0 && k0 + BK < K) {
            int nxt = cur ^ 1;
            tdm_load_tile(A + (size_t)rowBase * K + k0 + BK, ldsA_off + nxt * abuf, BM, K);
            tdm_load_tile(B + (size_t)colBase * K + k0 + BK, ldsB_off + nxt * bbuf, BN, K);
            if (FUSED)
                tdm_load_tile(B + (size_t)(IFF + colBase) * K + k0 + BK,
                              ldsB2_off + nxt * bbuf, BN, K);
        }
        const bf16* a  = lA + cur * (BM * LDT);
        const bf16* b  = lB + cur * (BN * LDT);
#else
    for (int k0 = 0; k0 < K; k0 += BK) {
        if (k0 + BK < K) {
            __builtin_prefetch(A + (size_t)(rowBase + (tid >> 1)) * K + (k0 + BK), 0, 1);
            __builtin_prefetch(B + (size_t)(colBase + (tid >> 2)) * K + (k0 + BK), 0, 1);
        }
#pragma unroll
        for (int pp = 0; pp < 2; ++pp) {
            int lin = pp * 256 + tid;
            int row = lin >> 2;
            int kc = (lin & 3) * 8;
            *(u32x4*)(lA + row * LDT + kc) =
                *(const u32x4*)(A + (size_t)(rowBase + row) * K + k0 + kc);
        }
        {
            int row = tid >> 2;
            int kc = (tid & 3) * 8;
            *(u32x4*)(lB + row * LDT + kc) =
                *(const u32x4*)(B + (size_t)(colBase + row) * K + k0 + kc);
            if (FUSED)
                *(u32x4*)(lB2 + row * LDT + kc) =
                    *(const u32x4*)(B + (size_t)(IFF + colBase + row) * K + k0 + kc);
        }
        __syncthreads();
        const bf16* a  = lA;
        const bf16* b  = lB;
#endif
        Frag fa[2], fb[2], fb2[2];
        // A frag (16x32 MxK): lanes 0-15 -> K [0,8)+[16,24), lanes 16-31 -> [8,16)+[24,32)
#pragma unroll
        for (int i = 0; i < 2; ++i) {
            int r = waveM * 32 + i * 16 + lhalf;
            int kb = hi * 8;
            fa[i].q[0] = *(const u32x4*)(a + r * LDT + kb);
            fa[i].q[1] = *(const u32x4*)(a + r * LDT + 16 + kb);
        }
        // B frag (32x16 KxN): lanes 0-15 -> K [0,8)+[16,24), lanes 16-31 -> [8,16)+[24,32)
#pragma unroll
        for (int j = 0; j < 2; ++j) {
            int c = waveN * 32 + j * 16 + lhalf;
            int kb = hi * 8;
            fb[j].q[0] = *(const u32x4*)(b + c * LDT + kb);
            fb[j].q[1] = *(const u32x4*)(b + c * LDT + kb + 16);
        }
        if (FUSED) {
#ifdef HAVE_TDM
            const bf16* b2 = lB2 + cur * (BN * LDT);
#else
            const bf16* b2 = lB2;
#endif
#pragma unroll
            for (int j = 0; j < 2; ++j) {
                int c = waveN * 32 + j * 16 + lhalf;
                int kb = hi * 8;
                fb2[j].q[0] = *(const u32x4*)(b2 + c * LDT + kb);
                fb2[j].q[1] = *(const u32x4*)(b2 + c * LDT + kb + 16);
            }
        }
#pragma unroll
        for (int i = 0; i < 2; ++i)
#pragma unroll
            for (int j = 0; j < 2; ++j)
                acc[i][j] = __builtin_amdgcn_wmma_f32_16x16x32_bf16(
                    false, fa[i].v, false, fb[j].v, (short)0, acc[i][j], false, false);
        if (FUSED) {
#pragma unroll
            for (int i = 0; i < 2; ++i)
#pragma unroll
                for (int j = 0; j < 2; ++j)
                    acc2[i][j] = __builtin_amdgcn_wmma_f32_16x16x32_bf16(
                        false, fa[i].v, false, fb2[j].v, (short)0, acc2[i][j], false, false);
        }
#ifdef HAVE_TDM
        if (wid == 0) __builtin_amdgcn_s_wait_tensorcnt(0);
        __syncthreads();
        cur ^= 1;
#else
        __syncthreads();
#endif
    }

    // epilogue: C/D layout -> vgpr r: M = r + 8*hi, N = lhalf
#pragma unroll
    for (int i = 0; i < 2; ++i)
#pragma unroll
        for (int j = 0; j < 2; ++j)
#pragma unroll
            for (int r = 0; r < 8; ++r) {
                int grow = rowBase + waveM * 32 + i * 16 + r + hi * 8;
                int gcol = colBase + waveN * 32 + j * 16 + lhalf;
                size_t idx = (size_t)grow * N + gcol;
                float v = acc[i][j][r];
                if (MODE == 0)      Cf[idx] = v;
                else if (MODE == 1) Cb[idx] = (bf16)v;
                else if (MODE == 2) Cf[idx] += v;
                else {
                    float u = acc2[i][j][r];
                    Cb[idx] = (bf16)((v / (1.f + expf(-v))) * u);
                }
            }
}

// ---------------------------------------------------------------------------
// attention: per (n, kv) block of 64 threads, loops over 8 query groups.
// RoPE on the fly (qkv bf16), S=9 softmax in fp32, writes bf16 O.
// ---------------------------------------------------------------------------
__global__ void attn_kernel(const bf16* __restrict__ qkv,
                            const float* __restrict__ cosq, const float* __restrict__ sinq,
                            bf16* __restrict__ O) {
    int n = blockIdx.x >> 1;
    int kv = blockIdx.x & 1;
    int d = threadIdx.x;             // 0..63
    int dp = d ^ HHALF;              // rotate-half partner
    __shared__ float sq[S_LEN][HD];
    __shared__ float sk[S_LEN][HD];
    __shared__ float sv[S_LEN][HD];
    __shared__ float sattn[S_LEN][S_LEN];

    for (int s = 0; s < S_LEN; ++s) {
        size_t base = ((size_t)n * S_LEN + s) * DQKV;
        float kval  = (float)qkv[base + DMODEL + kv * HD + d];
        float kpart = (float)qkv[base + DMODEL + kv * HD + dp];
        sk[s][d] = kval * cosq[s * HD + d] + kpart * sinq[s * HD + d];
        sv[s][d] = (float)qkv[base + DMODEL + KVH * HD + kv * HD + d];
    }
    for (int g = 0; g < G_HEADS; ++g) {
        __syncthreads();
        int qoff = (kv * G_HEADS + g) * HD;
        for (int s = 0; s < S_LEN; ++s) {
            size_t base = ((size_t)n * S_LEN + s) * DQKV;
            float qval  = (float)qkv[base + qoff + d];
            float qpart = (float)qkv[base + qoff + dp];
            sq[s][d] = qval * cosq[s * HD + d] + qpart * sinq[s * HD + d];
        }
        __syncthreads();
        for (int p = d; p < S_LEN * S_LEN; p += HD) {
            int s = p / S_LEN, t = p % S_LEN;
            float dot = 0.f;
#pragma unroll 8
            for (int k = 0; k < HD; ++k) dot += sq[s][k] * sk[t][k];
            sattn[s][t] = dot;
        }
        __syncthreads();
        if (d < S_LEN) {
            float mx = sattn[d][0];
            for (int t = 1; t < S_LEN; ++t) mx = fmaxf(mx, sattn[d][t]);
            float e[S_LEN];
            float sum = 0.f;
            for (int t = 0; t < S_LEN; ++t) { e[t] = expf(sattn[d][t] - mx); sum += e[t]; }
            float inv = 1.f / sum;
            for (int t = 0; t < S_LEN; ++t) sattn[d][t] = e[t] * inv;
        }
        __syncthreads();
        for (int s = 0; s < S_LEN; ++s) {
            float acc = 0.f;
#pragma unroll
            for (int t = 0; t < S_LEN; ++t) acc += sattn[s][t] * sv[t][d];
            O[((size_t)n * S_LEN + s) * DMODEL + qoff + d] = (bf16)acc;
        }
    }
}

// ---------------------------------------------------------------------------
// host-side launcher
// ---------------------------------------------------------------------------
extern "C" void kernel_launch(void* const* d_in, const int* in_sizes, int n_in,
                              void* d_out, int out_size, void* d_ws, size_t ws_size,
                              hipStream_t stream) {
    (void)in_sizes; (void)n_in; (void)out_size; (void)ws_size;
    const float* audio   = (const float*)d_in[0];
    const float* special = (const float*)d_in[1];
    const float* inW     = (const float*)d_in[2];
    const float* inB     = (const float*)d_in[3];
    const float* qkvW    = (const float*)d_in[4];
    const float* oW      = (const float*)d_in[5];
    const float* guW     = (const float*)d_in[6];
    const float* dnW     = (const float*)d_in[7];
    const float* lmW     = (const float*)d_in[8];
    const float* cosq    = (const float*)d_in[9];
    const float* sinq    = (const float*)d_in[10];
    float* out = (float*)d_out;

    char* ws = (char*)d_ws;
    size_t off = 0;
    auto alloc = [&](size_t bytes) -> void* {
        off = (off + 255) & ~(size_t)255;
        void* p = ws + off;
        off += bytes;
        return p;
    };

    const size_t qkvW_e = (size_t)LAYERS * DQKV * DMODEL;
    const size_t oW_e   = (size_t)LAYERS * DMODEL * DMODEL;
    const size_t guW_e  = (size_t)LAYERS * 2 * IFF * DMODEL;
    const size_t dnW_e  = (size_t)LAYERS * DMODEL * IFF;
    const size_t lmW_e  = (size_t)LMV * DMODEL;

    bf16*  qkvWb = (bf16*)alloc(qkvW_e * 2);
    bf16*  oWb   = (bf16*)alloc(oW_e * 2);
    bf16*  guWb  = (bf16*)alloc(guW_e * 2);
    bf16*  dnWb  = (bf16*)alloc(dnW_e * 2);
    bf16*  lmWb  = (bf16*)alloc(lmW_e * 2);
    float* H     = (float*)alloc((size_t)MTOK * DMODEL * 4);
    bf16*  Xn    = (bf16*) alloc((size_t)MTOK * DMODEL * 2);
    bf16*  QKVb  = (bf16*) alloc((size_t)MTOK * DQKV * 2);
    bf16*  Obuf  = (bf16*) alloc((size_t)MTOK * DMODEL * 2);
    bf16*  ACT   = (bf16*) alloc((size_t)MTOK * IFF * 2);
    bf16*  Feat  = (bf16*) alloc((size_t)N_SEQ * DMODEL * 2);

    // 1. weights -> bf16
    f2bf_kernel<<<2048, 256, 0, stream>>>(qkvW, qkvWb, qkvW_e);
    f2bf_kernel<<<2048, 256, 0, stream>>>(oW,   oWb,   oW_e);
    f2bf_kernel<<<4096, 256, 0, stream>>>(guW,  guWb,  guW_e);
    f2bf_kernel<<<4096, 256, 0, stream>>>(dnW,  dnWb,  dnW_e);
    f2bf_kernel<<<1024, 256, 0, stream>>>(lmW,  lmWb,  lmW_e);

    // 2. embed
    in_proj_kernel<<<N_SEQ * P_AUD, 256, 0, stream>>>(audio, inW, inB, H);
    copy_special_kernel<<<8192, 256, 0, stream>>>(special, H);

    // 3. layers
    for (int l = 0; l < LAYERS; ++l) {
        const bf16* qw = qkvWb + (size_t)l * DQKV * DMODEL;
        const bf16* ow = oWb   + (size_t)l * DMODEL * DMODEL;
        const bf16* gw = guWb  + (size_t)l * 2 * IFF * DMODEL;
        const bf16* dw = dnWb  + (size_t)l * DMODEL * IFF;

        rmsnorm_kernel<<<MTOK, 256, 0, stream>>>(H, DMODEL, Xn);
        gemm_bf16_wmma<1><<<dim3(DQKV / BN, MTOK / BM), 256, 0, stream>>>(
            Xn, qw, nullptr, QKVb, MTOK, DQKV, DMODEL);
        attn_kernel<<<N_SEQ * KVH, 64, 0, stream>>>(QKVb, cosq, sinq, Obuf);
        gemm_bf16_wmma<2><<<dim3(DMODEL / BN, MTOK / BM), 256, 0, stream>>>(
            Obuf, ow, H, nullptr, MTOK, DMODEL, DMODEL);

        rmsnorm_kernel<<<MTOK, 256, 0, stream>>>(H, DMODEL, Xn);
        // fused gate_up + silu*up -> ACT (bf16), no GU round-trip
        gemm_bf16_wmma<3><<<dim3(IFF / BN, MTOK / BM), 256, 0, stream>>>(
            Xn, gw, nullptr, ACT, MTOK, IFF, DMODEL);
        gemm_bf16_wmma<2><<<dim3(DMODEL / BN, MTOK / BM), 256, 0, stream>>>(
            ACT, dw, H, nullptr, MTOK, DMODEL, IFF);
    }

    // 4. head: rms of h[:,0] then lm projection
    rmsnorm_kernel<<<N_SEQ, 256, 0, stream>>>(H, (size_t)S_LEN * DMODEL, Feat);
    gemm_bf16_wmma<0><<<dim3(LMV / BN, N_SEQ / BM), 256, 0, stream>>>(
        Feat, lmWb, out, nullptr, N_SEQ, LMV, DMODEL);
}
